// MultiHeadAttention_28441273434816
// MI455X (gfx1250) — matrix operand order repaired and checked by reading, not compiled, not run on hardware
//
#include <hip/hip_runtime.h>
#include <hip/hip_bf16.h>

typedef __bf16 bf16;
typedef __attribute__((ext_vector_type(16))) __bf16 v16bf;
typedef __attribute__((ext_vector_type(8)))  __bf16 v8bf;
typedef __attribute__((ext_vector_type(8)))  float  v8f;
typedef __attribute__((ext_vector_type(4)))  unsigned int u32x4;
typedef __attribute__((ext_vector_type(8)))  unsigned int u32x8;

#define NHEADS  8
#define DMODEL  512
#define DQ      64
#define BATCH   4
#define SEQ     2048
#define MTOT    (BATCH * SEQ)   /* 8192 */

// ---------------------------------------------------------------------------
// WMMA fragment helpers (CDNA5 16x16x32 bf16, wave32).
// A-frag (16x32, MxK): lane L holds row M=L%16; K chunks [0..7],[16..23] for
// lanes 0-15 and [8..15],[24..31] for lanes 16-31 (ISA 7.12.2).
// B-frag (32x16, KxN) loaded from B^T stored row-major [n][k]: lane L holds
// col N=L%16 with the same K chunks as the A-frag ([0..7],[16..23] / [8..15],[24..31]).
// C-frag (16x16 f32): VGPR r <-> M = r + (lane<16 ? 0 : 8), N = lane%16.
// ---------------------------------------------------------------------------
static __device__ __forceinline__ v16bf load_a_frag(const bf16* base, int lda) {
  const int lane = threadIdx.x & 31;
  const bf16* p = base + (lane & 15) * lda + ((lane < 16) ? 0 : 8);
  v8bf lo = *(const v8bf*)(p);
  v8bf hi = *(const v8bf*)(p + 16);
  return __builtin_shufflevector(lo, hi, 0,1,2,3,4,5,6,7,8,9,10,11,12,13,14,15);
}

static __device__ __forceinline__ v16bf load_bt_frag(const bf16* baseT, int ldb) {
  const int lane = threadIdx.x & 31;
  const bf16* p = baseT + (lane & 15) * ldb + ((lane < 16) ? 0 : 8);
  v8bf lo = *(const v8bf*)(p);
  v8bf hi = *(const v8bf*)(p + 16);
  return __builtin_shufflevector(lo, hi, 0,1,2,3,4,5,6,7,8,9,10,11,12,13,14,15);
}

static __device__ __forceinline__ v8f wmma_bf16(v16bf a, v16bf b, v8f c) {
  return __builtin_amdgcn_wmma_f32_16x16x32_bf16(false, a, false, b,
                                                 (short)0, c, false, false);
}

static __device__ __forceinline__ v8f vzero8f() {
  v8f z = {0.f, 0.f, 0.f, 0.f, 0.f, 0.f, 0.f, 0.f};
  return z;
}

// ---------------------------------------------------------------------------
// Tensor Data Mover: DMA a 2D tile (rows x cols of bf16) from global to LDS.
// Builds the D# (ISA 8.3/8.4): group0 = {count=1, lds_addr, global_addr,
// type=2}; group1 = {data_size=2B, tensor dims, tile dims, dim0 stride}.
// Groups 2/3 zero-filled (2D). Issued wave-uniform; tracked by TENSORcnt.
// All dims/strides in elements; addresses in bytes.
// ---------------------------------------------------------------------------
static __device__ __forceinline__ void tdm_load_2d(unsigned int lds_addr,
                                                   const void* gaddr,
                                                   unsigned int tile_d0,
                                                   unsigned int tile_d1,
                                                   unsigned int tensor_d0,
                                                   unsigned int tensor_d1,
                                                   unsigned int stride0) {
  unsigned long long ga = (unsigned long long)gaddr;
  u32x4 g0 = { 1u,                                   // count=1, no gather
               lds_addr,                             // D#.lds_addr
               (unsigned int)ga,                     // global_addr[31:0]
               (unsigned int)((ga >> 32) & 0x01FFFFFFu) | (2u << 30) }; // +type=2
  u32x8 g1 = { 0x10000u,                             // data_size=1 (2 bytes)
               (tensor_d0 & 0xFFFFu) << 16,          // tensor_dim0[15:0]
               ((tensor_d0 >> 16) & 0xFFFFu) | ((tensor_d1 & 0xFFFFu) << 16),
               ((tensor_d1 >> 16) & 0xFFFFu) | (tile_d0 << 16),
               tile_d1 & 0xFFFFu,                    // tile_dim1 (tile_dim2=0)
               stride0,                              // tensor_dim0_stride[31:0]
               0u, 0u };
  u32x4 gz = { 0u, 0u, 0u, 0u };
  asm volatile("tensor_load_to_lds %0, %1, %2, %3"
               :
               : "s"(g0), "s"(g1), "s"(gz), "s"(gz)
               : "memory");
}

// ---------------------------------------------------------------------------
// Prep kernels: f32 -> bf16 conversion and weight re-layout.
// ---------------------------------------------------------------------------
__global__ void cvt_f32_to_bf16(const float* __restrict__ in,
                                bf16* __restrict__ out, int n) {
  int i = blockIdx.x * blockDim.x + threadIdx.x;
  if (i < n) out[i] = (bf16)in[i];
}

// W: [H][DMODEL][DQ] f32  ->  WT: [N=512][K=512] bf16, WT[n][k] = W[n/64][k][n%64]
__global__ void pack_headw_T(const float* __restrict__ W, bf16* __restrict__ WT) {
  int i = blockIdx.x * blockDim.x + threadIdx.x;   // over 512*512
  int n = i >> 9;
  int k = i & 511;
  WT[i] = (bf16)W[(((n >> 6) * DMODEL) + k) * DQ + (n & 63)];
}

// ---------------------------------------------------------------------------
// Generic [8192 x 512] = A[8192 x 512] * B (B^T given row-major [512][512]).
// Block = 4 waves; wave computes a 64(M) x 32(N) tile; block tile 64 x 128.
// MODE 0: bf16 out scattered to heads layout  [B][H][S][DQ]
// MODE 1: bf16 out transposed to vT layout    [B][H][DQ][S]  (packed b128 store)
// MODE 2: f32 out row-major (final d_out)
// ---------------------------------------------------------------------------
template <int MODE>
__global__ void __launch_bounds__(128) gemm512(const bf16* __restrict__ A,
                                               const bf16* __restrict__ BT,
                                               void* __restrict__ outp) {
  const int wave = threadIdx.x >> 5;
  const int lane = threadIdx.x & 31;
  const int m0 = blockIdx.x * 64;
  const int n0 = blockIdx.y * 128 + wave * 32;

  v8f c[4][2];
#pragma unroll
  for (int i = 0; i < 4; ++i)
#pragma unroll
    for (int j = 0; j < 2; ++j) c[i][j] = vzero8f();

#pragma unroll 4
  for (int k0 = 0; k0 < 512; k0 += 32) {
    if (k0 + 32 < 512) {
      __builtin_prefetch(A + (m0 + (lane & 15) * 4) * 512 + k0 + 32, 0, 1);
    }
    v16bf a[4];
#pragma unroll
    for (int i = 0; i < 4; ++i)
      a[i] = load_a_frag(A + (size_t)(m0 + i * 16) * 512 + k0, 512);
    v16bf b[2];
#pragma unroll
    for (int j = 0; j < 2; ++j)
      b[j] = load_bt_frag(BT + (size_t)(n0 + j * 16) * 512 + k0, 512);
#pragma unroll
    for (int i = 0; i < 4; ++i)
#pragma unroll
      for (int j = 0; j < 2; ++j) c[i][j] = wmma_bf16(a[i], b[j], c[i][j]);
  }

  const int nIdx = lane & 15;
  const int moff = (lane < 16) ? 0 : 8;
#pragma unroll
  for (int i = 0; i < 4; ++i) {
#pragma unroll
    for (int j = 0; j < 2; ++j) {
      const int n = n0 + j * 16 + nIdx;
      if (MODE == 2) {
        float* out = (float*)outp;
#pragma unroll
        for (int r = 0; r < 8; ++r) {
          int m = m0 + i * 16 + moff + r;
          out[(size_t)m * 512 + n] = c[i][j][r];
        }
      } else if (MODE == 0) {
        bf16* out = (bf16*)outp;
        const int h = n >> 6, q = n & 63;
#pragma unroll
        for (int r = 0; r < 8; ++r) {
          int m = m0 + i * 16 + moff + r;
          int bb = m >> 11, s = m & 2047;
          out[((size_t)(bb * NHEADS + h) * SEQ + s) * DQ + q] = (bf16)c[i][j][r];
        }
      } else {  // MODE 1: v transposed, 8 consecutive s per lane -> one b128
        bf16* out = (bf16*)outp;
        const int h = n >> 6, q = n & 63;
        const int m = m0 + i * 16 + moff;
        const int bb = m >> 11, s = m & 2047;
        v8bf pk;
#pragma unroll
        for (int r = 0; r < 8; ++r) pk[r] = (bf16)c[i][j][r];
        *(v8bf*)(out + ((size_t)(bb * NHEADS + h) * DQ + q) * SEQ + s) = pk;
      }
    }
  }
}

// ---------------------------------------------------------------------------
// Flash attention: grid (B*H, SEQ/64), block = 4 waves.
// K/V 64x64 tiles are DMA'd into LDS by the Tensor Data Mover (wave 0 issues,
// double-buffered, TENSORcnt-synchronized, barrier-published); all 4 waves
// read B-fragments from LDS instead of re-loading from global (4x VMEM cut).
// Wave owns 16 q-rows x dq=64; online softmax state per lane maps 1:1 to the
// C-fragment row layout. P re-shaped C->A via per-wave LDS staging.
// ---------------------------------------------------------------------------
__global__ void __launch_bounds__(128) flash_attn(const bf16* __restrict__ qh,
                                                  const bf16* __restrict__ kh,
                                                  const bf16* __restrict__ vt,
                                                  bf16* __restrict__ concat) {
  __shared__ __align__(16) bf16 ktile[2][64][64];  // [buf][t_local][kappa] 16KB
  __shared__ __align__(16) bf16 vtile[2][64][64];  // [buf][q][t_local]     16KB
  __shared__ __align__(16) bf16 plds[4][16][64];   // per-wave P staging     8KB

  const int wave = threadIdx.x >> 5;
  const int lane = threadIdx.x & 31;
  const int bh = blockIdx.x;
  const int b = bh >> 3, h = bh & 7;
  const bf16* qb = qh + (size_t)bh * SEQ * DQ;
  const bf16* kb = kh + (size_t)bh * SEQ * DQ;
  const bf16* vb = vt + (size_t)bh * DQ * SEQ;
  const int s0 = blockIdx.y * 64 + wave * 16;

  v16bf aq[2];
  aq[0] = load_a_frag(qb + (size_t)s0 * DQ, DQ);
  aq[1] = load_a_frag(qb + (size_t)s0 * DQ + 32, DQ);

  v8f o[4];
#pragma unroll
  for (int nf = 0; nf < 4; ++nf) o[nf] = vzero8f();
  float mrun[8], lrun[8];
#pragma unroll
  for (int r = 0; r < 8; ++r) { mrun[r] = -1e30f; lrun[r] = 0.f; }

  const int nIdx = lane & 15;
  const int moff = (lane < 16) ? 0 : 8;

  // prologue: DMA tile 0 into buffer 0
  if (wave == 0) {
    tdm_load_2d((unsigned int)(unsigned long long)&ktile[0][0][0],
                kb, 64, 64, DQ, SEQ, DQ);
    tdm_load_2d((unsigned int)(unsigned long long)&vtile[0][0][0],
                vb, 64, 64, SEQ, DQ, SEQ);
  }

  const int NT = SEQ / 64;  // 32 key tiles
  for (int it = 0; it < NT; ++it) {
    const int cur = it & 1;
    if (wave == 0) {
      if (it + 1 < NT) {  // DMA next tile into the other buffer
        const int t1 = (it + 1) * 64;
        tdm_load_2d((unsigned int)(unsigned long long)&ktile[1 - cur][0][0],
                    kb + (size_t)t1 * DQ, 64, 64, DQ, SEQ, DQ);
        tdm_load_2d((unsigned int)(unsigned long long)&vtile[1 - cur][0][0],
                    vb + t1, 64, 64, SEQ, DQ, SEQ);
        __builtin_amdgcn_s_wait_tensorcnt(2);  // in-order: tile `it` resident
      } else {
        __builtin_amdgcn_s_wait_tensorcnt(0);
      }
    }
    __syncthreads();  // publish tile `it` to all waves

    // S tile (16 x 64) = Q(16x64) * K^T(64x64), K from LDS
    v8f sfr[4];
#pragma unroll
    for (int nf = 0; nf < 4; ++nf) sfr[nf] = vzero8f();
#pragma unroll
    for (int ks = 0; ks < 2; ++ks) {
#pragma unroll
      for (int nf = 0; nf < 4; ++nf) {
        v16bf kf = load_bt_frag(&ktile[cur][nf * 16][ks * 32], 64);
        sfr[nf] = wmma_bf16(aq[ks], kf, sfr[nf]);
      }
    }
    // scale by 1/sqrt(dq)
#pragma unroll
    for (int nf = 0; nf < 4; ++nf)
#pragma unroll
      for (int r = 0; r < 8; ++r) sfr[nf][r] *= 0.125f;

    // per-row max across the 64 columns (4 frags x 16 lanes in half-wave)
    float mloc[8];
#pragma unroll
    for (int r = 0; r < 8; ++r)
      mloc[r] = fmaxf(fmaxf(sfr[0][r], sfr[1][r]), fmaxf(sfr[2][r], sfr[3][r]));
#pragma unroll
    for (int msk = 1; msk < 16; msk <<= 1)
#pragma unroll
      for (int r = 0; r < 8; ++r)
        mloc[r] = fmaxf(mloc[r], __shfl_xor(mloc[r], msk, 32));

    float alpha[8], rsum[8];
#pragma unroll
    for (int r = 0; r < 8; ++r) {
      float mn = fmaxf(mrun[r], mloc[r]);
      alpha[r] = __expf(mrun[r] - mn);
      mrun[r] = mn;
      rsum[r] = 0.f;
    }
    // P = exp(S - m); row sums
#pragma unroll
    for (int nf = 0; nf < 4; ++nf)
#pragma unroll
      for (int r = 0; r < 8; ++r) {
        float p = __expf(sfr[nf][r] - mrun[r]);
        sfr[nf][r] = p;
        rsum[r] += p;
      }
#pragma unroll
    for (int msk = 1; msk < 16; msk <<= 1)
#pragma unroll
      for (int r = 0; r < 8; ++r) rsum[r] += __shfl_xor(rsum[r], msk, 32);
#pragma unroll
    for (int r = 0; r < 8; ++r) lrun[r] = lrun[r] * alpha[r] + rsum[r];
    // rescale accumulators (row scale matches C-frag row mapping)
#pragma unroll
    for (int nf = 0; nf < 4; ++nf)
#pragma unroll
      for (int r = 0; r < 8; ++r) o[nf][r] *= alpha[r];

    // C-frag -> row-major bf16 P tile in per-wave LDS, reload as A-frags
#pragma unroll
    for (int nf = 0; nf < 4; ++nf)
#pragma unroll
      for (int r = 0; r < 8; ++r)
        plds[wave][moff + r][nf * 16 + nIdx] = (bf16)sfr[nf][r];

    v16bf pa[2];
    pa[0] = load_a_frag(&plds[wave][0][0], 64);
    pa[1] = load_a_frag(&plds[wave][0][0] + 32, 64);

    // O += P(16x64) * V(64x64), V^T tile from LDS (vtile[q][t_local])
#pragma unroll
    for (int ks = 0; ks < 2; ++ks) {
#pragma unroll
      for (int nf = 0; nf < 4; ++nf) {
        v16bf vf = load_bt_frag(&vtile[cur][nf * 16][ks * 32], 64);
        o[nf] = wmma_bf16(pa[ks], vf, o[nf]);
      }
    }
    __syncthreads();  // all waves done with tile `it` (buffer reuse at it+2)
  }

  // normalize + store in the reference's reshape layout:
  // concat[b][h*256 + s/8][(s%8)*64 + q]
#pragma unroll
  for (int nf = 0; nf < 4; ++nf) {
#pragma unroll
    for (int r = 0; r < 8; ++r) {
      float val = o[nf][r] / lrun[r];
      int s = s0 + moff + r;
      int q = nf * 16 + nIdx;
      size_t idx = (((size_t)b * SEQ) + (size_t)h * 256 + (s >> 3)) * 512 +
                   (size_t)(s & 7) * 64 + q;
      concat[idx] = (bf16)val;
    }
  }
}

// ---------------------------------------------------------------------------
// Host-side orchestration.
// Inputs (dict order): x_query, x_key, x_value, Q, K, V, Wout  (all f32)
// Output: [B, S, D_MODEL] f32
// ---------------------------------------------------------------------------
extern "C" void kernel_launch(void* const* d_in, const int* in_sizes, int n_in,
                              void* d_out, int out_size, void* d_ws, size_t ws_size,
                              hipStream_t stream) {
  (void)in_sizes; (void)n_in; (void)out_size; (void)ws_size;
  const float* xq   = (const float*)d_in[0];
  const float* xk   = (const float*)d_in[1];
  const float* xv   = (const float*)d_in[2];
  const float* Qw   = (const float*)d_in[3];
  const float* Kw   = (const float*)d_in[4];
  const float* Vw   = (const float*)d_in[5];
  const float* Wout = (const float*)d_in[6];
  float* out = (float*)d_out;

  const size_t XN = (size_t)MTOT * DMODEL;   // 8192*512 = 4,194,304 elems
  const size_t WN = (size_t)DMODEL * DMODEL; // 262,144 elems

  char* ws = (char*)d_ws;
  bf16* Xq   = (bf16*)ws; ws += XN * 2;
  bf16* Xk   = (bf16*)ws; ws += XN * 2;
  bf16* Xv   = (bf16*)ws; ws += XN * 2;
  bf16* WqT  = (bf16*)ws; ws += WN * 2;
  bf16* WkT  = (bf16*)ws; ws += WN * 2;
  bf16* WvT  = (bf16*)ws; ws += WN * 2;
  bf16* Wo   = (bf16*)ws; ws += WN * 2;   // B^T for out-proj == Wout row-major
  bf16* qhb  = (bf16*)ws; ws += XN * 2;   // [B][H][S][DQ]
  bf16* khb  = (bf16*)ws; ws += XN * 2;   // [B][H][S][DQ]
  bf16* vtb  = (bf16*)ws; ws += XN * 2;   // [B][H][DQ][S]
  bf16* conc = (bf16*)ws; ws += XN * 2;   // [B][S][DMODEL] ("buggy" reshape layout)

  // --- prep: conversions / re-layout ---
  cvt_f32_to_bf16<<<(unsigned)((XN + 255) / 256), 256, 0, stream>>>(xq, Xq, (int)XN);
  cvt_f32_to_bf16<<<(unsigned)((XN + 255) / 256), 256, 0, stream>>>(xk, Xk, (int)XN);
  cvt_f32_to_bf16<<<(unsigned)((XN + 255) / 256), 256, 0, stream>>>(xv, Xv, (int)XN);
  cvt_f32_to_bf16<<<(unsigned)((WN + 255) / 256), 256, 0, stream>>>(Wout, Wo, (int)WN);
  pack_headw_T<<<(unsigned)(WN / 256), 256, 0, stream>>>(Qw, WqT);
  pack_headw_T<<<(unsigned)(WN / 256), 256, 0, stream>>>(Kw, WkT);
  pack_headw_T<<<(unsigned)(WN / 256), 256, 0, stream>>>(Vw, WvT);

  // --- projections (WMMA bf16 GEMMs) ---
  dim3 gg(MTOT / 64, DMODEL / 128), gb(128);
  gemm512<0><<<gg, gb, 0, stream>>>(Xq, WqT, (void*)qhb);
  gemm512<0><<<gg, gb, 0, stream>>>(Xk, WkT, (void*)khb);
  gemm512<1><<<gg, gb, 0, stream>>>(Xv, WvT, (void*)vtb);

  // --- fused flash attention (TDM-staged K/V tiles) ---
  flash_attn<<<dim3(BATCH * NHEADS, SEQ / 64), 128, 0, stream>>>(qhb, khb, vtb, conc);

  // --- output projection ---
  gemm512<2><<<gg, gb, 0, stream>>>(conc, Wo, (void*)out);
}
